// FQGU_68092411511521
// MI455X (gfx1250) — compile-verified, matrix operand order checked by reading, not run on hardware
//
#include <hip/hip_runtime.h>
#include <hip/hip_bf16.h>

typedef __bf16 bf16;
typedef __attribute__((ext_vector_type(16))) __bf16 v16bf;
typedef __attribute__((ext_vector_type(8)))  float  v8f;

// ---------------------------------------------------------------------------
// 1x1 conv as GEMM:  Y[n,co,hw] = sum_ci W[co,ci] * X[n,ci,hw] + bias[co]
// One wave -> 16(M=Cout) x 32(N=hw) tile (two 16x16 WMMA tiles sharing A).
// HW is a template constant so all B loads use immediate VGLOBAL offsets
// (j*HW*4 <= 1.5MB < 8MB 24-bit limit) and batch into s_clause groups.
// A layout (16-bit 16x32, ISA 7.12.2): lanes 0-15 row M=lane, K {0..7,16..23};
// lanes 16-31 row M=lane-16, K {8..15,24..31}. B symmetric (columns of X).
// ---------------------------------------------------------------------------
template <int HW>
__global__ void __launch_bounds__(256)
conv1x1_wmma(const float* __restrict__ X, const float* __restrict__ W,
             const float* __restrict__ bias, float* __restrict__ Y,
             int Cin, int Cout)
{
    const int n    = blockIdx.x;
    const int mt   = blockIdx.y;
    const int lane = threadIdx.x & 31;
    const int wave = threadIdx.x >> 5;
    const int n0   = (blockIdx.z * 8 + wave) * 32;     // 32 output columns/wave
    const int col  = lane & 15;
    const int halfk = (lane & 16) ? 8 : 0;
    const int m    = mt * 16 + col;                    // output channel (A row)

    const float* Ap = W + (long)m * Cin + halfk;
    const float* Bp = X + (long)n * Cin * HW + (long)halfk * HW + n0 + col;

    v8f acc0 = {}, acc1 = {};
    for (int k0 = 0; k0 < Cin; k0 += 32) {
        if (k0 + 32 < Cin) __builtin_prefetch(Bp + (long)32 * HW, 0, 1);
        v16bf a, b0, b1;
#pragma unroll
        for (int j = 0; j < 8; ++j) {
            a[j]     = (bf16)Ap[j];
            a[j + 8] = (bf16)Ap[16 + j];
            b0[j]     = (bf16)Bp[(long)j * HW];
            b0[j + 8] = (bf16)Bp[(long)(16 + j) * HW];
            b1[j]     = (bf16)Bp[(long)j * HW + 16];
            b1[j + 8] = (bf16)Bp[(long)(16 + j) * HW + 16];
        }
        acc0 = __builtin_amdgcn_wmma_f32_16x16x32_bf16(false, a, false, b0,
                                                       (short)0, acc0, false, false);
        acc1 = __builtin_amdgcn_wmma_f32_16x16x32_bf16(false, a, false, b1,
                                                       (short)0, acc1, false, false);
        Ap += 32;
        Bp += (long)32 * HW;
    }
    const int rbase = (lane & 16) ? 8 : 0;
    float* Yp = Y + ((long)n * Cout + mt * 16 + rbase) * HW + n0 + col;
#pragma unroll
    for (int r = 0; r < 8; ++r) {
        const float bb = bias[mt * 16 + rbase + r];
        Yp[(long)r * HW]      = acc0[r] + bb;
        Yp[(long)r * HW + 16] = acc1[r] + bb;
    }
}

// ---------------------------------------------------------------------------
// 3x3 conv (pad=1) as implicit GEMM with K order (p,q)-major:
//   kflat = (p*3+q)*Cin + ci,  Cin = 96 = 3*32  =>  each 32-wide K-step has a
//   single fixed tap (p,q) -> fixed spatial base pointer, B loads at constant
//   ci*HW*4 offsets (<= 5.7MB, fits 24-bit).  W[co][ci][p][q] is gathered with
//   stride-9 constant offsets.  Padded M rows (Cout=9/25 -> 16/32) use a
//   clamped weight-row pointer; their garbage results are never stored
//   (rows of D are independent in D = A x B).
// ---------------------------------------------------------------------------
template <int Hd, int Wd, int Cout>
__global__ void __launch_bounds__(256)
conv3x3_wmma(const float* __restrict__ X, const float* __restrict__ W,
             const float* __restrict__ bias, float* __restrict__ Y, int Cin)
{
    constexpr int HW = Hd * Wd;
    const int n    = blockIdx.x;
    const int mt   = blockIdx.y;
    const int lane = threadIdx.x & 31;
    const int wave = threadIdx.x >> 5;
    const int n0   = (blockIdx.z * 8 + wave) * 32;
    const int col  = lane & 15;
    const int halfk = (lane & 16) ? 8 : 0;
    const int m     = mt * 16 + col;
    const int mrow  = (m < Cout) ? m : (Cout - 1);   // clamp; stores are masked
    // n0 is a multiple of 32 and Wd is a multiple of 32, so the 32 columns
    // [n0, n0+32) of this wave lie in one image row.
    const int hw0 = n0 + col;
    const int h0  = hw0 / Wd;
    const int w0  = hw0 - h0 * Wd;

    const float* Xn   = X + (long)n * Cin * HW;
    const float* Wrow = W + (long)mrow * Cin * 9;

    v8f acc0 = {}, acc1 = {};
#pragma unroll
    for (int p = 0; p < 3; ++p) {
        const int  hh = h0 + p - 1;
        const bool vh = (hh >= 0) && (hh < Hd);
        const int  hc = vh ? hh : 0;
#pragma unroll
        for (int q = 0; q < 3; ++q) {
            const int  pq  = p * 3 + q;
            const int  ww0 = w0 + q - 1;
            const int  ww1 = ww0 + 16;
            const bool v0  = vh && (ww0 >= 0) && (ww0 < Wd);
            const bool v1  = vh && (ww1 < Wd);          // ww1 >= 15 always
            const int  wc0 = v0 ? ww0 : 0;
            const int  wc1 = v1 ? ww1 : 0;
            const float* Bq0 = Xn + (long)hc * Wd + wc0 + (long)halfk * HW;
            const float* Bq1 = Xn + (long)hc * Wd + wc1 + (long)halfk * HW;
            const float* Ap  = Wrow + (long)halfk * 9 + pq;
#pragma unroll
            for (int c0 = 0; c0 < 96; c0 += 32) {       // Cin = 96: 3 steps
                v16bf a, b0, b1;
#pragma unroll
                for (int j = 0; j < 8; ++j) {
                    a[j]     = (bf16)Ap[j * 9];
                    a[j + 8] = (bf16)Ap[(16 + j) * 9];
                    const float x0a = Bq0[(long)j * HW];
                    const float x0b = Bq0[(long)(16 + j) * HW];
                    const float x1a = Bq1[(long)j * HW];
                    const float x1b = Bq1[(long)(16 + j) * HW];
                    b0[j]     = (bf16)(v0 ? x0a : 0.0f);
                    b0[j + 8] = (bf16)(v0 ? x0b : 0.0f);
                    b1[j]     = (bf16)(v1 ? x1a : 0.0f);
                    b1[j + 8] = (bf16)(v1 ? x1b : 0.0f);
                }
                acc0 = __builtin_amdgcn_wmma_f32_16x16x32_bf16(false, a, false, b0,
                                                               (short)0, acc0, false, false);
                acc1 = __builtin_amdgcn_wmma_f32_16x16x32_bf16(false, a, false, b1,
                                                               (short)0, acc1, false, false);
                Ap  += 32 * 9;
                Bq0 += (long)32 * HW;
                Bq1 += (long)32 * HW;
            }
        }
    }
    const int rbase = (lane & 16) ? 8 : 0;
    float* Yp = Y + ((long)n * Cout + mt * 16 + rbase) * HW + n0 + col;
#pragma unroll
    for (int r = 0; r < 8; ++r) {
        const int co = mt * 16 + rbase + r;
        if (co < Cout) {
            Yp[(long)r * HW]      = acc0[r] + bias[co];
            Yp[(long)r * HW + 16] = acc1[r] + bias[co];
        }
    }
}

// ---------------------------------------------------------------------------
// kernel_normalizer: out = exp(m - max)*ham / sum(exp(m - max)*ham)
// (softmax denominator cancels after the hamming renormalization).
// ---------------------------------------------------------------------------
template <int KK>
__global__ void __launch_bounds__(256)
knorm_kernel(const float* __restrict__ Min, const float* __restrict__ ham,
             float* __restrict__ Out, int HW, int total)
{
    const int pid = blockIdx.x * blockDim.x + threadIdx.x;
    if (pid >= total) return;
    const int  n    = pid / HW;
    const int  hw   = pid - n * HW;
    const long base = (long)n * KK * HW + hw;
    float v[KK];
    float mx = -3.4e38f;
#pragma unroll
    for (int i = 0; i < KK; ++i) { v[i] = Min[base + (long)i * HW]; mx = fmaxf(mx, v[i]); }
    float s = 0.0f;
#pragma unroll
    for (int i = 0; i < KK; ++i) { v[i] = __expf(v[i] - mx) * ham[i]; s += v[i]; }
    const float inv = 1.0f / s;
#pragma unroll
    for (int i = 0; i < KK; ++i) Out[base + (long)i * HW] = v[i] * inv;
}

// ---------------------------------------------------------------------------
// CARAFE (group=1) with LDS-staged masks:
// one block per (n, ho, 64-wide wo strip); the K*K x 64 mask values are loaded
// into LDS once and reused across all C channels (C up to 256 -> 256x fewer
// mask global loads).  Threads = 4 channels x 64 pixels.
// MODE 0: out = acc   MODE 1: out = addin + acc   MODE 2: out = 2x - acc
// ---------------------------------------------------------------------------
template <int K, int UP, int MODE>
__global__ void __launch_bounds__(256)
carafe_kernel(const float* __restrict__ X, const float* __restrict__ Msk,
              const float* __restrict__ AddIn, float* __restrict__ Out,
              int C, int Hi, int Wi)
{
    constexpr int KK = K * K;
    constexpr int r  = (K - 1) / 2;
    const int Ho  = Hi * UP, Wo = Wi * UP;
    const int n   = blockIdx.x;
    const int ho  = blockIdx.y;
    const int wo0 = blockIdx.z * 64;

    __shared__ float smask[KK * 64];
    const int tid = threadIdx.x;
    for (int i = tid; i < KK * 64; i += 256) {
        const int kk = i >> 6, px = i & 63;
        smask[i] = Msk[(((long)n * KK + kk) * Ho + ho) * Wo + wo0 + px];
    }
    __syncthreads();

    const int pix = tid & 63;
    const int c0  = tid >> 6;                 // 0..3
    const int wo  = wo0 + pix;
    const int xi  = ho / UP, yj = wo / UP;

    for (int c = c0; c < C; c += 4) {
        const float* Xc = X + ((long)n * C + c) * Hi * Wi;
        float acc = 0.0f;
#pragma unroll
        for (int p = 0; p < K; ++p) {
            const int hh = xi + p - r;
            if (hh < 0 || hh >= Hi) continue;
#pragma unroll
            for (int q = 0; q < K; ++q) {
                const int ww = yj + q - r;
                if (ww < 0 || ww >= Wi) continue;
                acc += Xc[(long)hh * Wi + ww] * smask[(p * K + q) * 64 + pix];
            }
        }
        const long oidx = (((long)n * C + c) * Ho + ho) * Wo + wo;
        float out;
        if (MODE == 0)      out = acc;
        else if (MODE == 1) out = AddIn[oidx] + acc;
        else                out = 2.0f * Xc[(long)xi * Wi + yj] - acc;  // UP==1
        Out[oidx] = out;
    }
}

// ---------------------------------------------------------------------------
extern "C" void kernel_launch(void* const* d_in, const int* in_sizes, int n_in,
                              void* d_out, int out_size, void* d_ws, size_t ws_size,
                              hipStream_t stream)
{
    const float* hr    = (const float*)d_in[0];
    const float* lr    = (const float*)d_in[1];
    const float* Whrc  = (const float*)d_in[2];
    const float* bhrc  = (const float*)d_in[3];
    const float* Wlrc  = (const float*)d_in[4];
    const float* blrc  = (const float*)d_in[5];
    const float* Wenc  = (const float*)d_in[6];
    const float* benc  = (const float*)d_in[7];
    const float* Wenc2 = (const float*)d_in[8];
    const float* benc2 = (const float*)d_in[9];
    const float* Wproj = (const float*)d_in[10];
    const float* bproj = (const float*)d_in[11];
    const float* hamL  = (const float*)d_in[12];   // 5x5
    const float* hamH  = (const float*)d_in[13];   // 3x3

    const int N = 4, C = 256, Cc = 96, Hh = 128, Wh = 128, Hl = 64, Wl = 64;
    const int HWh = Hh * Wh, HWl = Hl * Wl;

    // workspace layout (floats); peak ~88 MB, fits in L2 (192 MB)
    float* ws = (float*)d_ws;
    float* A  = ws;                          // c_hr           (N,Cc,128,128)
    float* E  = A  + (long)N * Cc * HWh;     // refined c_hr
    float* B  = E  + (long)N * Cc * HWh;     // c_lr           (N,Cc,64,64)
    float* Cm = B  + (long)N * Cc * HWl;     // m_hh           (N,9,128,128)
    float* D  = Cm + (long)N * 9  * HWh;     // mask_hr_init
    float* F  = D  + (long)N * 9  * HWh;     // m_lh           (N,25,128,128)
    float* G  = F  + (long)N * 25 * HWh;     // mask_lr_init
    float* H  = G  + (long)N * 25 * HWh;     // m_ll           (N,25,64,64)
    float* I  = H  + (long)N * 25 * HWl;     // mask_lr
    float* J  = I  + (long)N * 25 * HWh;     // m_hl           (N,9,64,64)
    float* Kb = J  + (long)N * 9  * HWl;     // mask_hr
    float* L  = Kb + (long)N * 9  * HWh;     // mask_hr_n
    float* Mb = ws;                          // lr_up (N,C,128,128); reuses dead front

    float* out_mask = (float*)d_out;                       // mask_lr_n (N,25,128,128)
    float* out_hr   = out_mask + (long)N * 25 * HWh;       // hr_out    (N,C,128,128)
    float* out_lr   = out_hr   + (long)N * C  * HWh;       // lr_out    (N,Cc,128,128)

    // 1. c_hr = conv1x1(hr, W_hrc)
    conv1x1_wmma<16384><<<dim3(N, Cc / 16, HWh / 256), 256, 0, stream>>>(hr, Whrc, bhrc, A, C, Cc);
    // 2. c_lr = conv1x1(lr, W_lrc)
    conv1x1_wmma<4096><<<dim3(N, Cc / 16, HWl / 256), 256, 0, stream>>>(lr, Wlrc, blrc, B, C, Cc);
    // 3. m_hh = conv3x3(c_hr, W_enc2)
    conv3x3_wmma<128, 128, 9><<<dim3(N, 1, HWh / 256), 256, 0, stream>>>(A, Wenc2, benc2, Cm, Cc);
    // 4. mask_hr_init = knorm(m_hh, 3)
    knorm_kernel<9><<<(N * HWh + 255) / 256, 256, 0, stream>>>(Cm, hamH, D, HWh, N * HWh);
    // 5. c_hr' = 2*c_hr - carafe(c_hr, mask_hr_init, k=3, up=1)
    carafe_kernel<3, 1, 2><<<dim3(N, Hh, Wh / 64), 256, 0, stream>>>(A, D, nullptr, E, Cc, Hh, Wh);
    // 6. m_lh = conv3x3(c_hr', W_enc)
    conv3x3_wmma<128, 128, 25><<<dim3(N, 2, HWh / 256), 256, 0, stream>>>(E, Wenc, benc, F, Cc);
    // 7. mask_lr_init = knorm(m_lh, 5)
    knorm_kernel<25><<<(N * HWh + 255) / 256, 256, 0, stream>>>(F, hamL, G, HWh, N * HWh);
    // 8. m_ll = conv3x3(c_lr, W_enc)
    conv3x3_wmma<64, 64, 25><<<dim3(N, 2, HWl / 256), 256, 0, stream>>>(B, Wenc, benc, H, Cc);
    // 9. mask_lr = m_lh + carafe(m_ll, mask_lr_init, k=5, up=2)   (resize = identity)
    carafe_kernel<5, 2, 1><<<dim3(N, Hh, Wh / 64), 256, 0, stream>>>(H, G, F, I, 25, Hl, Wl);
    // 10. mask_lr_n = knorm(mask_lr, 5)  (== mask_lr_init2; computed once, reused)
    knorm_kernel<25><<<(N * HWh + 255) / 256, 256, 0, stream>>>(I, hamL, out_mask, HWh, N * HWh);
    // 11. m_hl = conv3x3(c_lr, W_enc2)
    conv3x3_wmma<64, 64, 9><<<dim3(N, 1, HWl / 256), 256, 0, stream>>>(B, Wenc2, benc2, J, Cc);
    // 12. mask_hr = m_hh + carafe(m_hl, mask_lr_n, k=5, up=2)
    carafe_kernel<5, 2, 1><<<dim3(N, Hh, Wh / 64), 256, 0, stream>>>(J, out_mask, Cm, Kb, 9, Hl, Wl);
    // 13. mask_hr_n = knorm(mask_hr, 3)
    knorm_kernel<9><<<(N * HWh + 255) / 256, 256, 0, stream>>>(Kb, hamH, L, HWh, N * HWh);
    // 14. hr_out = 2*hr - carafe(hr, mask_hr_n, k=3, up=1)
    carafe_kernel<3, 1, 2><<<dim3(N, Hh, Wh / 64), 256, 0, stream>>>(hr, L, nullptr, out_hr, C, Hh, Wh);
    // 15. lr_up = carafe(lr, mask_lr_n, k=5, up=2)   (front of ws is dead now)
    carafe_kernel<5, 2, 0><<<dim3(N, Hh, Wh / 64), 256, 0, stream>>>(lr, out_mask, nullptr, Mb, C, Hl, Wl);
    // 16. lr_out = conv1x1(lr_up, W_proj)
    conv1x1_wmma<16384><<<dim3(N, Cc / 16, HWh / 256), 256, 0, stream>>>(Mb, Wproj, bproj, out_lr, C, Cc);
}
